// UpsampleUpdatingModel2_78151224918750
// MI455X (gfx1250) — matrix operand order repaired and checked by reading, not compiled, not run on hardware
//
#include <hip/hip_runtime.h>
#include <hip/hip_bf16.h>

typedef _Float16 h16;
typedef __attribute__((ext_vector_type(16))) _Float16 v16h;
typedef __attribute__((ext_vector_type(8)))  _Float16 v8h;
typedef __attribute__((ext_vector_type(8)))  float    v8f;
typedef __attribute__((ext_vector_type(4)))  unsigned int u32x4;

#define H56 56
#define CH  256
#define NIMG 24
#define HW  (H56*H56)

// ---------------------------------------------------------------------------
// 1) bilinear 112->56 resize of motions, times scale 0.25.
//    half-pixel sampling at 2x reduction == plain 2x2 average.
__global__ void resize_pm_kernel(const float* __restrict__ pin, float* __restrict__ pout) {
    int idx = blockIdx.x * 256 + threadIdx.x;           // 24*2*56*56
    if (idx >= NIMG * 2 * HW) return;
    int x = idx % H56, y = (idx / H56) % H56, nc = idx / HW;
    const float* s = pin + (size_t)nc * 112 * 112;
    int y0 = 2 * y, x0 = 2 * x;
    float v = 0.25f * (s[y0*112 + x0] + s[y0*112 + x0 + 1] +
                       s[(y0+1)*112 + x0] + s[(y0+1)*112 + x0 + 1]);
    pout[idx] = v * 0.25f;                              // * scale (56/224)
}

// ---------------------------------------------------------------------------
// 2) weight repack f32 OIHW -> f16 [tap][cchunk][o][32]  (3x3, Cin=256)
__global__ void wcvt3x3_kernel(const float* __restrict__ w, h16* __restrict__ out) {
    int idx = blockIdx.x * 256 + threadIdx.x;           // 9*8*256*32 = 589824
    if (idx >= 9 * 8 * CH * 32) return;
    int cl = idx & 31, o = (idx >> 5) & 255, cc = (idx >> 13) & 7, t = idx >> 16;
    int c = cc * 32 + cl;
    out[idx] = (h16)w[((size_t)o * CH + c) * 9 + t];
}
// dw repack f32 [o][c] -> f16 [cchunk][o][32]
__global__ void wcvtdw_kernel(const float* __restrict__ w, h16* __restrict__ out) {
    int idx = blockIdx.x * 256 + threadIdx.x;           // 65536
    if (idx >= 8 * CH * 32) return;
    int cl = idx & 31, o = (idx >> 5) & 255, cc = idx >> 13;
    out[idx] = (h16)w[(size_t)o * CH + cc * 32 + cl];
}

// ---------------------------------------------------------------------------
// 3) tiny conv: 2ch motions -> 256ch, ReLU, f16 NHWC out
__global__ __launch_bounds__(256)
void conv_ew1_kernel(const float* __restrict__ pm, const float* __restrict__ w,
                     const float* __restrict__ b, h16* __restrict__ dst) {
    int x = blockIdx.x, y = blockIdx.y, n = blockIdx.z;
    __shared__ float patch[18];
    int t = threadIdx.x;
    if (t < 18) {
        int c = t / 9, tap = t % 9;
        int gy = y + tap / 3 - 1, gx = x + tap % 3 - 1;
        float v = 0.f;
        if ((unsigned)gy < 56u && (unsigned)gx < 56u)
            v = pm[(((size_t)n * 2 + c) * H56 + gy) * H56 + gx];
        patch[t] = v;
    }
    __syncthreads();
    int o = threadIdx.x;
    float s = b[o];
    const float* wo = w + (size_t)o * 18;
#pragma unroll
    for (int i = 0; i < 18; ++i) s += wo[i] * patch[i];
    s = fmaxf(s, 0.f);
    dst[(((size_t)n * H56 + y) * H56 + x) * CH + o] = (h16)s;
}

// ---------------------------------------------------------------------------
// 4) deformable 1x1 bilinear sample of i_features (L2-resident) -> f16 NHWC
__global__ __launch_bounds__(256)
void deform_kernel(const float* __restrict__ feat, const float* __restrict__ pm,
                   h16* __restrict__ dst) {
    int x = blockIdx.x, y = blockIdx.y, n = blockIdx.z;
    float dy = pm[(((size_t)n * 2 + 0) * H56 + y) * H56 + x];
    float dx = pm[(((size_t)n * 2 + 1) * H56 + y) * H56 + x];
    float gy = (float)y + dy, gx = (float)x + dx;
    float y0f = floorf(gy), x0f = floorf(gx);
    float ty = gy - y0f, tx = gx - x0f;
    int c = threadIdx.x;
    const float* base = feat + ((size_t)(n / 3) * CH + c) * HW;
    float acc = 0.f;
#pragma unroll
    for (int q = 0; q < 4; ++q) {
        float yf = y0f + (float)(q >> 1), xf = x0f + (float)(q & 1);
        float wq = ((q >> 1) ? ty : 1.f - ty) * ((q & 1) ? tx : 1.f - tx);
        if (yf >= 0.f && yf < 56.f && xf >= 0.f && xf < 56.f)
            acc += wq * base[(int)yf * H56 + (int)xf];
    }
    dst[(((size_t)n * H56 + y) * H56 + x) * CH + c] = (h16)acc;
}

// ---------------------------------------------------------------------------
// 5) WMMA implicit-im2col conv: Cin=256 -> Cout=256, ksize 1 or 3, SAME pad.
//    src f16 NHWC, wgt f16 [tap][cchunk][256][32], per-WG 8x8 pixel tile.
//    8 waves x (2 M-tiles x 4 N-tiles) of v_wmma_f32_16x16x32_f16.
//    global_prefetch of the next weight slab overlaps L2 fetch with compute.
__global__ __launch_bounds__(256)
void conv_wmma_kernel(const h16* __restrict__ src, const h16* __restrict__ wgt,
                      const float* __restrict__ bias, h16* __restrict__ dst,
                      const h16* __restrict__ emb, float* __restrict__ p2d,
                      int ksize, int doRelu) {
    const int bx = blockIdx.x, by = blockIdx.y, n = blockIdx.z;
    const int pad  = ksize >> 1;
    const int PW   = 8 + 2 * pad;
    const int taps = ksize * ksize;
    const int tid  = threadIdx.x;
    const int lane = tid & 31;
    const int wv   = tid >> 5;
    const int hf   = lane >> 4;     // lane half (wave32)
    const int nl   = lane & 15;

    __shared__ __attribute__((aligned(32))) h16 patch[100 * 32];   // up to 10x10 halo, 32ch, 6.25KB
    __shared__ __attribute__((aligned(16))) h16 wlds[CH * 32];     // 16KB weight slab

    v8f zero = {0.f, 0.f, 0.f, 0.f, 0.f, 0.f, 0.f, 0.f};
    v8f acc[2][4];
#pragma unroll
    for (int i = 0; i < 2; ++i)
#pragma unroll
        for (int j = 0; j < 4; ++j) acc[i][j] = zero;

    const u32x4* wsrc = (const u32x4*)wgt;
    u32x4* wldsv  = (u32x4*)wlds;
    u32x4* patchv = (u32x4*)patch;

    for (int cc = 0; cc < 8; ++cc) {
        __syncthreads();                       // prior reads of patch/wlds done
        // stage input patch (zero-padded halo), f16 NHWC -> [pix][32ch]
        const int npix = PW * PW;
        for (int idx = tid; idx < npix * 4; idx += 256) {
            int p = idx >> 2, seg = idx & 3;
            int py = p / PW, px = p % PW;
            int gy = by * 8 + py - pad, gx = bx * 8 + px - pad;
            u32x4 v = {0u, 0u, 0u, 0u};
            if ((unsigned)gy < 56u && (unsigned)gx < 56u)
                v = *(const u32x4*)(src + ((((size_t)n * H56 + gy) * H56 + gx) * CH + cc * 32 + seg * 8));
            patchv[p * 4 + seg] = v;
        }
        for (int t = 0; t < taps; ++t) {
            if (t) __syncthreads();
            const u32x4* wsrc_t = wsrc + (size_t)(t * 8 + cc) * 1024;  // 16KB contiguous
#pragma unroll
            for (int q = 0; q < 4; ++q) wldsv[tid + q * 256] = wsrc_t[tid + q * 256];

            // prefetch next weight slab into L2/L0 while this stage computes
            {
                int nt = t + 1, ncc = cc;
                if (nt == taps) { nt = 0; ncc = cc + 1; }
                if (ncc < 8) {
                    const char* pnext = (const char*)(wsrc + (size_t)(nt * 8 + ncc) * 1024);
                    __builtin_prefetch(pnext + (size_t)tid * 64, 0, 3);   // 256 thr x 64B = 16KB
                }
            }
            __syncthreads();

            const int oy = t / ksize, ox = t % ksize;
            // B fragments: K = hf*16 + j (channel-minor, contiguous 32B)
            v16h bf[4];
#pragma unroll
            for (int j = 0; j < 4; ++j) {
                int pix = j * 16 + nl;
                int py = (pix >> 3) + oy, px = (pix & 7) + ox;
                bf[j] = *(const v16h*)(patch + (py * PW + px) * 32 + hf * 16);
            }
#pragma unroll
            for (int i = 0; i < 2; ++i) {
                int o = (wv * 2 + i) * 16 + nl;
                v8h alo = *(const v8h*)(wlds + o * 32 + hf * 16);       // K = hf*16 .. +7
                v8h ahi = *(const v8h*)(wlds + o * 32 + hf * 16 + 8);   // K = hf*16+8 .. +7
                v16h af = __builtin_shufflevector(alo, ahi,
                          0,1,2,3,4,5,6,7,8,9,10,11,12,13,14,15);
#pragma unroll
                for (int j = 0; j < 4; ++j)
                    acc[i][j] = __builtin_amdgcn_wmma_f32_16x16x32_f16(
                        false, af, false, bf[j], (short)0, acc[i][j], false, false);
            }
        }
        // prefetch next C-chunk's interior patch lines (8 rows x 512B each)
        if (cc < 7 && tid < 64) {
            int py = tid >> 3, seg8 = tid & 7;       // 8 rows x 8 segments of 64B
            int gy = by * 8 + py, gx = bx * 8;
            const char* pn = (const char*)(src + ((((size_t)n * H56 + gy) * H56 + gx) * CH + (cc + 1) * 32));
            __builtin_prefetch(pn + (size_t)seg8 * 64, 0, 3);
        }
    }

    // epilogue: bias, optional ReLU, f16 NHWC store and/or f32 NCHW p2d (+emb)
#pragma unroll
    for (int i = 0; i < 2; ++i) {
        int o0 = (wv * 2 + i) * 16 + hf * 8;          // 8 consecutive out channels
#pragma unroll
        for (int j = 0; j < 4; ++j) {
            int pix = j * 16 + nl;
            int gy = by * 8 + (pix >> 3), gx = bx * 8 + (pix & 7);
            v8f r = acc[i][j];
#pragma unroll
            for (int v = 0; v < 8; ++v) {
                float f = r[v] + bias[o0 + v];
                if (doRelu) f = fmaxf(f, 0.f);
                r[v] = f;
            }
            size_t pixbase = (((size_t)n * H56 + gy) * H56 + gx) * CH + o0;
            if (dst) {
                v8h hh;
#pragma unroll
                for (int v = 0; v < 8; ++v) hh[v] = (h16)r[v];
                *(v8h*)(dst + pixbase) = hh;
            }
            if (p2d) {
                v8h e = *(const v8h*)(emb + pixbase);
#pragma unroll
                for (int v = 0; v < 8; ++v)
                    p2d[((size_t)(n * CH + o0 + v)) * HW + gy * H56 + gx] = r[v] + (float)e[v];
            }
        }
    }
}

// ---------------------------------------------------------------------------
// 6) mean over h,w of p2d -> pooled[n][c]
__global__ __launch_bounds__(256)
void pool_kernel(const float* __restrict__ p2d, float* __restrict__ pooled) {
    int row = blockIdx.x;                         // n*256 + c, 6144 rows
    const float* s = p2d + (size_t)row * HW;
    float acc = 0.f;
    for (int i = threadIdx.x; i < HW; i += 256) acc += s[i];
    __shared__ float red[256];
    red[threadIdx.x] = acc;
    __syncthreads();
    for (int off = 128; off > 0; off >>= 1) {
        if ((int)threadIdx.x < off) red[threadIdx.x] += red[threadIdx.x + off];
        __syncthreads();
    }
    if (threadIdx.x == 0) pooled[row] = red[0] * (1.f / (float)HW);
}

// ---------------------------------------------------------------------------
extern "C" void kernel_launch(void* const* d_in, const int* in_sizes, int n_in,
                              void* d_out, int out_size, void* d_ws, size_t ws_size,
                              hipStream_t stream) {
    (void)in_sizes; (void)n_in; (void)out_size; (void)ws_size;
    const float* i_features = (const float*)d_in[1];
    const float* p_motions  = (const float*)d_in[2];
    const float* dw  = (const float*)d_in[3];
    const float* db  = (const float*)d_in[4];
    const float* mw1 = (const float*)d_in[5];  const float* mb1 = (const float*)d_in[6];
    const float* mw2 = (const float*)d_in[7];  const float* mb2 = (const float*)d_in[8];
    const float* mw3 = (const float*)d_in[9];  const float* mb3 = (const float*)d_in[10];
    const float* ew1 = (const float*)d_in[11]; const float* eb1 = (const float*)d_in[12];
    const float* ew2 = (const float*)d_in[13]; const float* eb2 = (const float*)d_in[14];
    const float* ew3 = (const float*)d_in[15]; const float* eb3 = (const float*)d_in[16];
    const float* ew4 = (const float*)d_in[17]; const float* eb4 = (const float*)d_in[18];

    float* pooled = (float*)d_out;
    float* p2d    = (float*)d_out + (size_t)NIMG * CH;      // 6144 offset

    // workspace layout (all 256B aligned)
    char* ws = (char*)d_ws;
    const size_t SZ_PM  = (size_t)NIMG * 2 * HW * sizeof(float);          // 602,112
    const size_t SZ_ACT = (size_t)NIMG * HW * CH * sizeof(h16);           // 38,535,168
    const size_t SZ_W33 = (size_t)9 * 8 * CH * 32 * sizeof(h16);          // 1,179,648
    float* pm   = (float*)(ws);
    h16* bufA   = (h16*)(ws + SZ_PM);
    h16* bufB   = (h16*)(ws + SZ_PM + SZ_ACT);
    h16* bufE   = (h16*)(ws + SZ_PM + 2 * SZ_ACT);
    h16* wbase  = (h16*)(ws + SZ_PM + 3 * SZ_ACT);
    h16* wE2 = wbase;                h16* wE3 = wbase + 1 * (SZ_W33/2);
    h16* wE4 = wbase + 2*(SZ_W33/2); h16* wM1 = wbase + 3 * (SZ_W33/2);
    h16* wM2 = wbase + 4*(SZ_W33/2); h16* wM3 = wbase + 5 * (SZ_W33/2);
    h16* wDW = wbase + 6*(SZ_W33/2);

    // motions resize + scale
    resize_pm_kernel<<<(NIMG*2*HW + 255)/256, 256, 0, stream>>>(p_motions, pm);

    // weight repacks (f32 -> f16 WMMA-friendly layout)
    int n33 = 9 * 8 * CH * 32;
    wcvt3x3_kernel<<<(n33+255)/256, 256, 0, stream>>>(ew2, wE2);
    wcvt3x3_kernel<<<(n33+255)/256, 256, 0, stream>>>(ew3, wE3);
    wcvt3x3_kernel<<<(n33+255)/256, 256, 0, stream>>>(ew4, wE4);
    wcvt3x3_kernel<<<(n33+255)/256, 256, 0, stream>>>(mw1, wM1);
    wcvt3x3_kernel<<<(n33+255)/256, 256, 0, stream>>>(mw2, wM2);
    wcvt3x3_kernel<<<(n33+255)/256, 256, 0, stream>>>(mw3, wM3);
    wcvtdw_kernel<<<(8*CH*32 + 255)/256, 256, 0, stream>>>(dw, wDW);

    dim3 pixgrid(H56, H56, NIMG);
    dim3 cgrid(7, 7, NIMG);

    // embedding branch: ew1 (direct) then three WMMA 3x3 convs
    conv_ew1_kernel<<<pixgrid, 256, 0, stream>>>(pm, ew1, eb1, bufA);
    conv_wmma_kernel<<<cgrid, 256, 0, stream>>>(bufA, wE2, eb2, bufB, nullptr, nullptr, 3, 1);
    conv_wmma_kernel<<<cgrid, 256, 0, stream>>>(bufB, wE3, eb3, bufA, nullptr, nullptr, 3, 1);
    conv_wmma_kernel<<<cgrid, 256, 0, stream>>>(bufA, wE4, eb4, bufE, nullptr, nullptr, 3, 1);

    // feature branch: deform sample, 1x1 WMMA matmul (no relu), three 3x3 convs
    deform_kernel<<<pixgrid, 256, 0, stream>>>(i_features, pm, bufA);
    conv_wmma_kernel<<<cgrid, 256, 0, stream>>>(bufA, wDW, db, bufB, nullptr, nullptr, 1, 0);
    conv_wmma_kernel<<<cgrid, 256, 0, stream>>>(bufB, wM1, mb1, bufA, nullptr, nullptr, 3, 1);
    conv_wmma_kernel<<<cgrid, 256, 0, stream>>>(bufA, wM2, mb2, bufB, nullptr, nullptr, 3, 1);
    // final conv fuses relu + emb residual, writes f32 NCHW p2d into d_out
    conv_wmma_kernel<<<cgrid, 256, 0, stream>>>(bufB, wM3, mb3, nullptr, bufE, p2d, 3, 1);

    // pooled mean over h,w
    pool_kernel<<<NIMG * CH, 256, 0, stream>>>(p2d, pooled);
}
